// MultiHeadAttention_6983616824391
// MI455X (gfx1250) — matrix operand order repaired and checked by reading, not compiled, not run on hardware
//
#include <hip/hip_runtime.h>

// ---------------- problem constants ----------------
constexpr int SEQ = 2048;
constexpr int DM  = 1024;
constexpr int NH  = 16;
constexpr int DK  = 64;
constexpr int NB  = 2;
constexpr int MT  = NB * SEQ;          // 4096 rows for the projection GEMMs
constexpr float QK_SCALE = 0.125f;     // 1/sqrt(64)
constexpr float MASK_VAL = -1e9f;

// ---------------- vector types ----------------
typedef __attribute__((ext_vector_type(16))) __bf16        v16bf;
typedef __attribute__((ext_vector_type(8)))  float         v8f;
typedef __attribute__((ext_vector_type(4)))  float         f32x4;
typedef __attribute__((ext_vector_type(4)))  unsigned int  u32x4;
typedef __attribute__((ext_vector_type(4)))  unsigned short u16x4;

union Frag {                 // one WMMA A/B operand: 16 bf16 = 32 bytes
    v16bf bf;
    u32x4 q[2];
    unsigned short u[16];
};

__device__ __forceinline__ unsigned short f2bf(float f) {
    unsigned int u = __builtin_bit_cast(unsigned int, f);
    u += 0x7FFFu + ((u >> 16) & 1u);      // round-to-nearest-even
    return (unsigned short)(u >> 16);
}

__device__ __forceinline__ v8f wmma_bf16(v16bf a, v16bf b, v8f c) {
    return __builtin_amdgcn_wmma_f32_16x16x32_bf16(
        /*neg_a=*/false, a, /*neg_b=*/false, b,
        /*c_mod=*/(short)0, c, /*reuse_a=*/false, /*reuse_b=*/false);
}

// ---- CDNA5 async global->LDS copy (16B per lane, tracked by ASYNCcnt) ----
// VDST = VGPR holding LDS byte offset; VADDR = 64-bit global address.
__device__ __forceinline__ void async_copy_b128(unsigned lds_off, const void* gptr) {
    unsigned long long ga = (unsigned long long)(uintptr_t)gptr;
    asm volatile("global_load_async_to_lds_b128 %0, %1, off"
                 :: "v"(lds_off), "v"(ga)
                 : "memory");
}
__device__ __forceinline__ void wait_asynccnt0() {
    asm volatile("s_wait_asynccnt 0x0" ::: "memory");
}

// LDS tile: 128 rows x 32 bf16, padded to 40 shorts (80 B, keeps 16B alignment)
constexpr int LDS_LD = 40;

// =====================================================================
// K1: P = X @ W^T + bias   (X fp32 [MT,DM], W fp32 [DM,DM]) -> bf16
//     v_transpose==1 stores per-head transposed layout (b,h,d,s) for V.
// 128x128 block tile, 8 waves of 32x64, K-step 32.
// =====================================================================
__global__ __launch_bounds__(256)
void proj_kernel(const float* __restrict__ X, const float* __restrict__ W,
                 const float* __restrict__ bias, unsigned short* __restrict__ Y,
                 int v_transpose) {
    __shared__ unsigned short As[128 * LDS_LD];
    __shared__ unsigned short Bs[128 * LDS_LD];

    const int tid  = threadIdx.x;
    const int lane = tid & 31, wid = tid >> 5;
    const int wm = wid >> 1, wn = wid & 1;          // wave grid 4(M) x 2(N)
    const int lm = lane & 15, lh = lane >> 4;
    const int n0 = blockIdx.x * 128, m0 = blockIdx.y * 128;

    v8f acc[2][4] = {};

    for (int k0 = 0; k0 < DM; k0 += 32) {
        // ---- stage 128x32 fp32 -> bf16 into LDS (A and B tiles) ----
        #pragma unroll
        for (int p = 0; p < 4; ++p) {
            const int r = (tid >> 3) + p * 32;
            const int c = (tid & 7) * 4;
            f32x4 av = *(const f32x4*)(X + (size_t)(m0 + r) * DM + k0 + c);
            f32x4 bv = *(const f32x4*)(W + (size_t)(n0 + r) * DM + k0 + c);
            u16x4 ap, bp;
            #pragma unroll
            for (int j = 0; j < 4; ++j) { ap[j] = f2bf(av[j]); bp[j] = f2bf(bv[j]); }
            *(u16x4*)(&As[r * LDS_LD + c]) = ap;
            *(u16x4*)(&Bs[r * LDS_LD + c]) = bp;
        }
        if (k0 + 32 < DM) {   // hint next K-slab into cache (global_prefetch_b8)
            __builtin_prefetch(X + (size_t)(m0 + (tid >> 3)) * DM + k0 + 32, 0, 0);
            __builtin_prefetch(W + (size_t)(n0 + (tid >> 3)) * DM + k0 + 32, 0, 0);
        }
        __syncthreads();

        // ---- fragments (ISA §7.12.2 layouts) ----
        Frag a[2], bb[4];
        #pragma unroll
        for (int mi = 0; mi < 2; ++mi) {
            const int row = wm * 32 + mi * 16 + lm;
            a[mi].q[0] = *(const u32x4*)(&As[row * LDS_LD + lh * 8]);
            a[mi].q[1] = *(const u32x4*)(&As[row * LDS_LD + 16 + lh * 8]);
        }
        #pragma unroll
        for (int ni = 0; ni < 4; ++ni) {
            const int nn = wn * 64 + ni * 16 + lm;
            bb[ni].q[0] = *(const u32x4*)(&Bs[nn * LDS_LD + lh * 8]);
            bb[ni].q[1] = *(const u32x4*)(&Bs[nn * LDS_LD + 16 + lh * 8]);
        }
        #pragma unroll
        for (int mi = 0; mi < 2; ++mi)
            #pragma unroll
            for (int ni = 0; ni < 4; ++ni)
                acc[mi][ni] = wmma_bf16(a[mi].bf, bb[ni].bf, acc[mi][ni]);
        __syncthreads();
    }

    // ---- epilogue: +bias, convert, store ----
    #pragma unroll
    for (int mi = 0; mi < 2; ++mi)
        #pragma unroll
        for (int ni = 0; ni < 4; ++ni) {
            const int col = n0 + wn * 64 + ni * 16 + lm;
            const float bv = bias[col];
            #pragma unroll
            for (int r = 0; r < 8; ++r) {
                const int row = m0 + wm * 32 + mi * 16 + lh * 8 + r;
                const float v = acc[mi][ni][r] + bv;
                if (v_transpose) {   // (b,h,d,s) layout for V
                    const int b = row >> 11, s = row & (SEQ - 1);
                    const int h = col >> 6, d = col & (DK - 1);
                    Y[(size_t)(((b << 4) + h) * DK + d) * SEQ + s] = f2bf(v);
                } else {
                    Y[(size_t)row * DM + col] = f2bf(v);
                }
            }
        }
}

// =====================================================================
// K2: scores = (Qh @ Kh^T) * 1/8, causal mask -> fp32 attn buffer.
// K=64 -> fragments straight from global bf16.
// =====================================================================
__global__ __launch_bounds__(256)
void scores_kernel(const unsigned short* __restrict__ Qb,
                   const unsigned short* __restrict__ Kb,
                   float* __restrict__ attn) {
    const int k0 = blockIdx.x * 128, q0 = blockIdx.y * 128;
    const int bh = blockIdx.z, b = bh >> 4, h = bh & 15;
    const int tid = threadIdx.x;
    const int lane = tid & 31, wid = tid >> 5;
    const int wm = wid >> 1, wn = wid & 1;
    const int lm = lane & 15, lh = lane >> 4;

    if (k0 > q0 + 127) {   // tile fully above diagonal: all masked
        #pragma unroll
        for (int mi = 0; mi < 2; ++mi)
            #pragma unroll
            for (int ni = 0; ni < 4; ++ni) {
                const int kc = k0 + wn * 64 + ni * 16 + lm;
                #pragma unroll
                for (int r = 0; r < 8; ++r) {
                    const int qr = q0 + wm * 32 + mi * 16 + lh * 8 + r;
                    attn[(size_t)(bh * SEQ + qr) * SEQ + kc] = MASK_VAL;
                }
            }
        return;
    }

    v8f acc[2][4] = {};
    #pragma unroll
    for (int step = 0; step < 2; ++step) {
        const int dco = h * DK + step * 32;
        Frag a[2], bb[4];
        #pragma unroll
        for (int mi = 0; mi < 2; ++mi) {
            const size_t base = (size_t)(b * SEQ + q0 + wm * 32 + mi * 16 + lm) * DM + dco;
            a[mi].q[0] = *(const u32x4*)(Qb + base + lh * 8);
            a[mi].q[1] = *(const u32x4*)(Qb + base + 16 + lh * 8);
        }
        #pragma unroll
        for (int ni = 0; ni < 4; ++ni) {
            const size_t base = (size_t)(b * SEQ + k0 + wn * 64 + ni * 16 + lm) * DM + dco + lh * 8;
            bb[ni].q[0] = *(const u32x4*)(Kb + base);
            bb[ni].q[1] = *(const u32x4*)(Kb + base + 16);
        }
        #pragma unroll
        for (int mi = 0; mi < 2; ++mi)
            #pragma unroll
            for (int ni = 0; ni < 4; ++ni)
                acc[mi][ni] = wmma_bf16(a[mi].bf, bb[ni].bf, acc[mi][ni]);
    }

    #pragma unroll
    for (int mi = 0; mi < 2; ++mi)
        #pragma unroll
        for (int ni = 0; ni < 4; ++ni) {
            const int kc = k0 + wn * 64 + ni * 16 + lm;
            #pragma unroll
            for (int r = 0; r < 8; ++r) {
                const int qr = q0 + wm * 32 + mi * 16 + lh * 8 + r;
                const float v = (kc <= qr) ? acc[mi][ni][r] * QK_SCALE : MASK_VAL;
                attn[(size_t)(bh * SEQ + qr) * SEQ + kc] = v;
            }
        }
}

// =====================================================================
// K3: row softmax in-place over attn (one 256-thread block per row).
// =====================================================================
__global__ __launch_bounds__(256)
void softmax_kernel(float* __restrict__ attn) {
    float* p = attn + (size_t)blockIdx.x * SEQ;
    const int t = threadIdx.x, lane = t & 31, wid = t >> 5;
    __shared__ float red[8];

    float x[8];
    #pragma unroll
    for (int i = 0; i < 8; ++i) x[i] = p[t + i * 256];

    float m = -3.4e38f;
    #pragma unroll
    for (int i = 0; i < 8; ++i) m = fmaxf(m, x[i]);
    #pragma unroll
    for (int off = 16; off > 0; off >>= 1) m = fmaxf(m, __shfl_xor(m, off, 32));
    if (lane == 0) red[wid] = m;
    __syncthreads();
    float M = red[0];
    #pragma unroll
    for (int j = 1; j < 8; ++j) M = fmaxf(M, red[j]);
    __syncthreads();

    float s = 0.f;
    #pragma unroll
    for (int i = 0; i < 8; ++i) { x[i] = __expf(x[i] - M); s += x[i]; }
    #pragma unroll
    for (int off = 16; off > 0; off >>= 1) s += __shfl_xor(s, off, 32);
    if (lane == 0) red[wid] = s;
    __syncthreads();
    float S = 0.f;
    #pragma unroll
    for (int j = 0; j < 8; ++j) S += red[j];
    const float inv = 1.0f / S;
    #pragma unroll
    for (int i = 0; i < 8; ++i) p[t + i * 256] = x[i] * inv;
}

// =====================================================================
// K4: ctx = attn @ Vh  (attn fp32 -> bf16 in regs; Vt bf16 (b,h,d,s)).
// 128(q) x 64(d) block tile, 8 waves of 32x32, K-step 32 over keys.
// ctx stored bf16 interleaved (b,s,h*64+d) so K5 reads it as [MT,DM].
// =====================================================================
__global__ __launch_bounds__(256)
void ctx_kernel(const float* __restrict__ attn,
                const unsigned short* __restrict__ Vt,
                unsigned short* __restrict__ Cb) {
    const int q0 = blockIdx.x * 128;
    const int bh = blockIdx.y, b = bh >> 4, h = bh & 15;
    const int tid = threadIdx.x;
    const int lane = tid & 31, wid = tid >> 5;
    const int wm = wid >> 1, wn = wid & 1;          // 4(M) x 2(N of 32)
    const int lm = lane & 15, lh = lane >> 4;

    v8f acc[2][2] = {};

    for (int step = 0; step < SEQ / 32; ++step) {
        Frag a[2], bb[2];
        #pragma unroll
        for (int mi = 0; mi < 2; ++mi) {
            const float* ap = attn + (size_t)(bh * SEQ + q0 + wm * 32 + mi * 16 + lm) * SEQ
                                   + step * 32;
            const f32x4 v0 = *(const f32x4*)(ap + lh * 8);
            const f32x4 v1 = *(const f32x4*)(ap + lh * 8 + 4);
            const f32x4 v2 = *(const f32x4*)(ap + 16 + lh * 8);
            const f32x4 v3 = *(const f32x4*)(ap + 16 + lh * 8 + 4);
            #pragma unroll
            for (int j = 0; j < 4; ++j) {
                a[mi].u[j]      = f2bf(v0[j]);
                a[mi].u[4 + j]  = f2bf(v1[j]);
                a[mi].u[8 + j]  = f2bf(v2[j]);
                a[mi].u[12 + j] = f2bf(v3[j]);
            }
        }
        #pragma unroll
        for (int ni = 0; ni < 2; ++ni) {
            const int d = wn * 32 + ni * 16 + lm;
            const size_t base = (size_t)(bh * DK + d) * SEQ + step * 32 + lh * 8;
            bb[ni].q[0] = *(const u32x4*)(Vt + base);
            bb[ni].q[1] = *(const u32x4*)(Vt + base + 16);
        }
        #pragma unroll
        for (int mi = 0; mi < 2; ++mi)
            #pragma unroll
            for (int ni = 0; ni < 2; ++ni)
                acc[mi][ni] = wmma_bf16(a[mi].bf, bb[ni].bf, acc[mi][ni]);
    }

    #pragma unroll
    for (int mi = 0; mi < 2; ++mi)
        #pragma unroll
        for (int ni = 0; ni < 2; ++ni) {
            const int col = h * DK + wn * 32 + ni * 16 + lm;
            #pragma unroll
            for (int r = 0; r < 8; ++r) {
                const int row = b * SEQ + q0 + wm * 32 + mi * 16 + lh * 8 + r;
                Cb[(size_t)row * DM + col] = f2bf(acc[mi][ni][r]);
            }
        }
}

// =====================================================================
// K5: out = ctx @ Wo^T + b_o  (ctx bf16 [MT,DM], Wo fp32) -> fp32 d_out
// A tile is a pure bf16 copy -> use GLOBAL_LOAD_ASYNC_TO_LDS_B128.
// =====================================================================
__global__ __launch_bounds__(256)
void oproj_kernel(const unsigned short* __restrict__ A, const float* __restrict__ W,
                  const float* __restrict__ bias, float* __restrict__ Y) {
    __shared__ unsigned short As[128 * LDS_LD];
    __shared__ unsigned short Bs[128 * LDS_LD];

    const int tid  = threadIdx.x;
    const int lane = tid & 31, wid = tid >> 5;
    const int wm = wid >> 1, wn = wid & 1;
    const int lm = lane & 15, lh = lane >> 4;
    const int n0 = blockIdx.x * 128, m0 = blockIdx.y * 128;

    // LDS byte offset of this thread's A-tile slot (low 32 bits of the
    // generic pointer are the LDS offset per the aperture mapping)
    const int ar = tid >> 1, ahalf = tid & 1;
    const unsigned a_lds_off =
        (unsigned)(uintptr_t)(void*)&As[ar * LDS_LD + ahalf * 16];

    v8f acc[2][4] = {};

    for (int k0 = 0; k0 < DM; k0 += 32) {
        {   // A tile: async DMA copy, 2 x 16B per thread, no VGPR round-trip
            const unsigned short* src = A + (size_t)(m0 + ar) * DM + k0 + ahalf * 16;
            async_copy_b128(a_lds_off,      src);
            async_copy_b128(a_lds_off + 16, src + 8);
        }
        #pragma unroll
        for (int p = 0; p < 4; ++p) {   // B tile: fp32 -> bf16
            const int r = (tid >> 3) + p * 32;
            const int c = (tid & 7) * 4;
            f32x4 bv = *(const f32x4*)(W + (size_t)(n0 + r) * DM + k0 + c);
            u16x4 bp;
            #pragma unroll
            for (int j = 0; j < 4; ++j) bp[j] = f2bf(bv[j]);
            *(u16x4*)(&Bs[r * LDS_LD + c]) = bp;
        }
        if (k0 + 32 < DM)
            __builtin_prefetch(W + (size_t)(n0 + (tid >> 3)) * DM + k0 + 32, 0, 0);
        wait_asynccnt0();                // our async copies have landed in LDS
        __syncthreads();

        Frag a[2], bb[4];
        #pragma unroll
        for (int mi = 0; mi < 2; ++mi) {
            const int row = wm * 32 + mi * 16 + lm;
            a[mi].q[0] = *(const u32x4*)(&As[row * LDS_LD + lh * 8]);
            a[mi].q[1] = *(const u32x4*)(&As[row * LDS_LD + 16 + lh * 8]);
        }
        #pragma unroll
        for (int ni = 0; ni < 4; ++ni) {
            const int nn = wn * 64 + ni * 16 + lm;
            bb[ni].q[0] = *(const u32x4*)(&Bs[nn * LDS_LD + lh * 8]);
            bb[ni].q[1] = *(const u32x4*)(&Bs[nn * LDS_LD + 16 + lh * 8]);
        }
        #pragma unroll
        for (int mi = 0; mi < 2; ++mi)
            #pragma unroll
            for (int ni = 0; ni < 4; ++ni)
                acc[mi][ni] = wmma_bf16(a[mi].bf, bb[ni].bf, acc[mi][ni]);
        __syncthreads();
    }

    #pragma unroll
    for (int mi = 0; mi < 2; ++mi)
        #pragma unroll
        for (int ni = 0; ni < 4; ++ni) {
            const int col = n0 + wn * 64 + ni * 16 + lm;
            const float bv = bias[col];
            #pragma unroll
            for (int r = 0; r < 8; ++r) {
                const int row = m0 + wm * 32 + mi * 16 + lh * 8 + r;
                Y[(size_t)row * DM + col] = acc[mi][ni][r] + bv;
            }
        }
}

// =====================================================================
extern "C" void kernel_launch(void* const* d_in, const int* in_sizes, int n_in,
                              void* d_out, int out_size, void* d_ws, size_t ws_size,
                              hipStream_t stream) {
    const float* q   = (const float*)d_in[0];
    const float* k   = (const float*)d_in[1];
    const float* v   = (const float*)d_in[2];
    // d_in[3] = causal mask (tril) — applied analytically in scores_kernel
    const float* w_q = (const float*)d_in[4];
    const float* b_q = (const float*)d_in[5];
    const float* w_k = (const float*)d_in[6];
    const float* b_k = (const float*)d_in[7];
    const float* w_v = (const float*)d_in[8];
    const float* b_v = (const float*)d_in[9];
    const float* w_o = (const float*)d_in[10];
    const float* b_o = (const float*)d_in[11];

    constexpr size_t PROJ_ELEMS = (size_t)MT * DM;   // 4,194,304
    unsigned short* Qb = (unsigned short*)d_ws;
    unsigned short* Kb = Qb + PROJ_ELEMS;
    unsigned short* Vt = Qb + 2 * PROJ_ELEMS;        // (b,h,d,s) layout
    unsigned short* Cb = Qb + 3 * PROJ_ELEMS;        // ctx, (b,s,h*64+d)

    float* out  = (float*)d_out;
    float* attn = out + PROJ_ELEMS;                  // (B,H,S,S) fp32

    const dim3 blk(256);
    const dim3 gproj(DM / 128, MT / 128);            // (8, 32)

    proj_kernel<<<gproj, blk, 0, stream>>>(q, w_q, b_q, Qb, 0);
    proj_kernel<<<gproj, blk, 0, stream>>>(k, w_k, b_k, Kb, 0);
    proj_kernel<<<gproj, blk, 0, stream>>>(v, w_v, b_v, Vt, 1);

    scores_kernel<<<dim3(SEQ / 128, SEQ / 128, NB * NH), blk, 0, stream>>>(Qb, Kb, attn);
    softmax_kernel<<<dim3(NB * NH * SEQ), blk, 0, stream>>>(attn);
    ctx_kernel<<<dim3(SEQ / 128, NB * NH), blk, 0, stream>>>(attn, Vt, Cb);
    oproj_kernel<<<gproj, blk, 0, stream>>>(Cb, w_o, b_o, out);
}
